// TemporalSegmentProposalNetwork_75806172774569
// MI455X (gfx1250) — matrix operand order repaired and checked by reading, not compiled, not run on hardware
//
#include <hip/hip_runtime.h>
#include <hip/hip_bf16.h>

typedef __attribute__((ext_vector_type(16))) __bf16 v16bf;
typedef __attribute__((ext_vector_type(8)))  float  v8f;

struct alignas(16) U4 { unsigned x, y, z, w; };
struct alignas(16) F4 { float x, y, z, w; };

union AB  { v16bf v; U4 q[2]; };
union OB  { __bf16 h[8]; U4 q; };

#define DD 2048
#define TT 256
#define PT 320      // padded T: 32 zero rows + 256 data + 32 zero rows
#define CC 20

__device__ __forceinline__ unsigned lds_addr(const void* p) {
  return (unsigned)(size_t)p;   // flat shared ptr low 32 bits = wave LDS offset
}

// ---------------------------------------------------------------------------
// Zero-fill bf16 region (pads + cw buffer) once per launch
// ---------------------------------------------------------------------------
__global__ void zero_bf16_kernel(__bf16* __restrict__ p, int n) {
  int idx = blockIdx.x * blockDim.x + threadIdx.x;
  if (idx < n) p[idx] = (__bf16)0.0f;
}

// ---------------------------------------------------------------------------
// Convert conv weight [O][I][3] f32 -> [tap][O][I] bf16
// ---------------------------------------------------------------------------
__global__ void convw_kernel(const float* __restrict__ W, __bf16* __restrict__ WB) {
  int idx = blockIdx.x * blockDim.x + threadIdx.x;   // o*DD + i
  const float* p = W + (size_t)idx * 3;
  float w0 = p[0], w1 = p[1], w2 = p[2];
  WB[idx]                        = (__bf16)w0;
  WB[(size_t)DD * DD + idx]      = (__bf16)w1;
  WB[(size_t)2 * DD * DD + idx]  = (__bf16)w2;
}

// Elementwise f32 -> bf16 (for cw rows 0..19; rows 20..31 stay zero)
__global__ void convcw_kernel(const float* __restrict__ S, __bf16* __restrict__ D, int n) {
  int idx = blockIdx.x * blockDim.x + threadIdx.x;
  if (idx < n) D[idx] = (__bf16)S[idx];
}

// ---------------------------------------------------------------------------
// MaxPool1d over T (-inf pad) on x [B][T][D] f32 -> bf16 [B][PT][D] (row t+32)
// ---------------------------------------------------------------------------
__global__ void maxpool_kernel(const float* __restrict__ X, __bf16* __restrict__ H, int pad) {
  int idx = blockIdx.x * blockDim.x + threadIdx.x;   // b*T*D + t*D + d
  int d = idx & (DD - 1);
  int t = (idx >> 11) & (TT - 1);
  int b = idx >> 19;
  float m = -3.402823466e38f;
  for (int w = -pad; w <= pad; ++w) {
    int tt = t + w;
    if (tt >= 0 && tt < TT) {
      float v = X[((size_t)b * TT + tt) * DD + d];
      m = v > m ? v : m;
    }
  }
  H[((size_t)b * PT + (t + 32)) * DD + d] = (__bf16)m;
}

// ---------------------------------------------------------------------------
// Dilated kernel-3 conv as 3 shifted GEMMs with bf16 WMMA + async-to-LDS
// double-buffered staging.
// W: [3][2048][2048] bf16, X: [B][PT][D] bf16 (zero-padded rows),
// bias f32[O]. Ybf: [B][PT][D] bf16. Yf32: logits [B][T][D] f32 or null.
// WG tile 128(M=O) x 64(N=T), 8 waves (4x2), each wave 2x2 WMMA 16x16.
// ---------------------------------------------------------------------------
__global__ __launch_bounds__(256)
void conv_gemm_kernel(const __bf16* __restrict__ W, const __bf16* __restrict__ X,
                      const float* __restrict__ bias, int dil,
                      __bf16* __restrict__ Ybf, float* __restrict__ Yf32) {
  __shared__ __bf16 lA[2][128][40];   // [buf][M][K] rows 80B
  __shared__ __bf16 lB[2][64][40];    // [buf][N(t)][K(i)]

  const int m0 = blockIdx.x * 128;
  const int n0 = blockIdx.y * 64;
  const int b  = blockIdx.z;
  const int tid  = threadIdx.x;
  const int lane = tid & 31;
  const int wave = tid >> 5;
  const int wm = wave & 3;
  const int wn = wave >> 2;
  const int hf  = lane >> 4;
  const int l16 = lane & 15;

  v8f zero;
  #pragma unroll
  for (int i = 0; i < 8; ++i) zero[i] = 0.0f;
  v8f acc[2][2];
  acc[0][0] = zero; acc[0][1] = zero; acc[1][0] = zero; acc[1][1] = zero;

  const int arow = tid >> 1, acol = (tid & 1) * 16;  // A: 2 thr/row, 16 elems
  const int brow = tid >> 2, bcol = (tid & 3) * 8;   // B: 4 thr/row, 8 elems
  const size_t Xb = (size_t)b * PT * DD;

  const int NCH = 3 * (DD / 32);                     // 192 chunks

  auto stage = [&](int buf, int kc) {
    int tap = kc >> 6;                               // DD/32 == 64 chunks per tap
    int kk  = (kc & 63) * 32;
    int shift = (tap - 1) * dil;
    const __bf16* ga = W + (size_t)tap * DD * DD + (size_t)(m0 + arow) * DD + kk + acol;
    unsigned la = lds_addr(&lA[buf][arow][acol]);
    asm volatile("global_load_async_to_lds_b128 %0, %1, off"
                 :: "v"(la), "v"(ga) : "memory");
    asm volatile("global_load_async_to_lds_b128 %0, %1, off offset:16"
                 :: "v"(la), "v"(ga) : "memory");
    int tg = n0 + brow + shift + 32;                 // padded row, always in range
    const __bf16* gb = X + Xb + (size_t)tg * DD + kk + bcol;
    unsigned lb = lds_addr(&lB[buf][brow][bcol]);
    asm volatile("global_load_async_to_lds_b128 %0, %1, off"
                 :: "v"(lb), "v"(gb) : "memory");
  };

  stage(0, 0);
  for (int kc = 0; kc < NCH; ++kc) {
    const int cur = kc & 1;
    asm volatile("s_wait_asynccnt 0x0" ::: "memory");
    __syncthreads();
    if (kc + 1 < NCH) stage(cur ^ 1, kc + 1);        // overlap DMA with WMMA

    AB afr[2], bfr[2];
    #pragma unroll
    for (int ms = 0; ms < 2; ++ms) {
      int row = wm * 32 + ms * 16 + l16;
      afr[ms].q[0] = *(const U4*)&lA[cur][row][hf * 8];
      afr[ms].q[1] = *(const U4*)&lA[cur][row][16 + hf * 8];
    }
    #pragma unroll
    for (int ns = 0; ns < 2; ++ns) {
      int row = wn * 32 + ns * 16 + l16;
      bfr[ns].q[0] = *(const U4*)&lB[cur][row][hf * 8];
      bfr[ns].q[1] = *(const U4*)&lB[cur][row][16 + hf * 8];
    }
    #pragma unroll
    for (int ms = 0; ms < 2; ++ms)
      #pragma unroll
      for (int ns = 0; ns < 2; ++ns)
        acc[ms][ns] = __builtin_amdgcn_wmma_f32_16x16x32_bf16(
            false, afr[ms].v, false, bfr[ns].v, (short)0, acc[ms][ns], false, false);
  }

  // epilogue: C layout VGPR r, lane half -> M = r + 8*hf ; N = l16
  #pragma unroll
  for (int ms = 0; ms < 2; ++ms) {
    const int obase = m0 + wm * 32 + ms * 16 + hf * 8;
    float bz[8];
    #pragma unroll
    for (int r = 0; r < 8; ++r) bz[r] = bias[obase + r];
    #pragma unroll
    for (int ns = 0; ns < 2; ++ns) {
      const int t = n0 + wn * 32 + ns * 16 + l16;
      float v[8];
      #pragma unroll
      for (int r = 0; r < 8; ++r) v[r] = acc[ms][ns][r] + bz[r];
      OB ob;
      #pragma unroll
      for (int r = 0; r < 8; ++r) ob.h[r] = (__bf16)v[r];
      *(U4*)(Ybf + Xb + (size_t)(t + 32) * DD + obase) = ob.q;
      if (Yf32) {
        F4 f0 = {v[0], v[1], v[2], v[3]};
        F4 f1 = {v[4], v[5], v[6], v[7]};
        F4* p = (F4*)(Yf32 + ((size_t)b * TT + t) * DD + obase);
        p[0] = f0; p[1] = f1;
      }
    }
  }
}

// ---------------------------------------------------------------------------
// Scoring GEMM: scores[b][t][c] = relu(cb[c] + sum_{j<Lwin} sum_d
//     cw[c][j*D+d] * logits[b][t-so+j][d]), windows read from padded LG.
// M = t (tile 128, 8 waves x 16 rows), N = c (32, 20 valid, rows 20..31 zero),
// K = Lwin*D. Async double-buffered staging.
// ---------------------------------------------------------------------------
__global__ __launch_bounds__(256)
void score_gemm_kernel(const __bf16* __restrict__ LG, const __bf16* __restrict__ CW,
                       const float* __restrict__ CB, int Lwin, int so,
                       float* __restrict__ SC) {
  __shared__ __bf16 lA[2][128][40];
  __shared__ __bf16 lB[2][32][40];

  const int t0 = blockIdx.x * 128;
  const int b  = blockIdx.y;
  const int tid  = threadIdx.x;
  const int lane = tid & 31;
  const int wave = tid >> 5;
  const int hf  = lane >> 4;
  const int l16 = lane & 15;
  const int mrow = wave * 16;

  v8f zero;
  #pragma unroll
  for (int i = 0; i < 8; ++i) zero[i] = 0.0f;
  v8f acc[2]; acc[0] = zero; acc[1] = zero;

  const int arow = tid >> 1, acol = (tid & 1) * 16;
  const int brow = tid >> 3, bcol = (tid & 7) * 4;   // 32 rows x 4 elems (8B)
  const size_t K2  = (size_t)Lwin * DD;
  const size_t LGb = (size_t)b * PT * DD;
  const int nK = Lwin * (DD / 32);

  auto stage = [&](int buf, int kc) {
    int gk = kc * 32;
    int j  = gk >> 11;
    int d0 = gk & (DD - 1);
    int tg = t0 + arow - so + j + 32;                // padded, always in range
    const __bf16* ga = LG + LGb + (size_t)tg * DD + d0 + acol;
    unsigned la = lds_addr(&lA[buf][arow][acol]);
    asm volatile("global_load_async_to_lds_b128 %0, %1, off"
                 :: "v"(la), "v"(ga) : "memory");
    asm volatile("global_load_async_to_lds_b128 %0, %1, off offset:16"
                 :: "v"(la), "v"(ga) : "memory");
    const __bf16* gb = CW + (size_t)brow * K2 + gk + bcol;
    unsigned lb = lds_addr(&lB[buf][brow][bcol]);
    asm volatile("global_load_async_to_lds_b64 %0, %1, off"
                 :: "v"(lb), "v"(gb) : "memory");
  };

  stage(0, 0);
  for (int kc = 0; kc < nK; ++kc) {
    const int cur = kc & 1;
    asm volatile("s_wait_asynccnt 0x0" ::: "memory");
    __syncthreads();
    if (kc + 1 < nK) stage(cur ^ 1, kc + 1);

    AB afr, bfr[2];
    {
      int row = mrow + l16;
      afr.q[0] = *(const U4*)&lA[cur][row][hf * 8];
      afr.q[1] = *(const U4*)&lA[cur][row][16 + hf * 8];
    }
    #pragma unroll
    for (int ns = 0; ns < 2; ++ns) {
      int crow = ns * 16 + l16;
      bfr[ns].q[0] = *(const U4*)&lB[cur][crow][hf * 8];
      bfr[ns].q[1] = *(const U4*)&lB[cur][crow][16 + hf * 8];
    }
    acc[0] = __builtin_amdgcn_wmma_f32_16x16x32_bf16(
        false, afr.v, false, bfr[0].v, (short)0, acc[0], false, false);
    acc[1] = __builtin_amdgcn_wmma_f32_16x16x32_bf16(
        false, afr.v, false, bfr[1].v, (short)0, acc[1], false, false);
  }

  #pragma unroll
  for (int ns = 0; ns < 2; ++ns) {
    const int c = ns * 16 + l16;
    if (c < CC) {
      const float cb = CB[c];
      #pragma unroll
      for (int r = 0; r < 8; ++r) {
        const int t = t0 + mrow + r + 8 * hf;
        float v = acc[ns][r] + cb;
        SC[((size_t)b * TT + t) * CC + c] = v > 0.0f ? v : 0.0f;
      }
    }
  }
}

// ---------------------------------------------------------------------------
// Stable bitonic sort of 256 (key=-sum_c score, idx=t) pairs per (f,b)
// ---------------------------------------------------------------------------
__global__ __launch_bounds__(256)
void sort_kernel(const float* __restrict__ SC, int* __restrict__ ORD) {
  __shared__ float key[256];
  __shared__ int   idx[256];
  const int fb = blockIdx.x, tid = threadIdx.x;
  const float* s = SC + ((size_t)fb * TT + tid) * CC;
  float sum = 0.0f;
  for (int c = 0; c < CC; ++c) sum += s[c];
  key[tid] = -sum; idx[tid] = tid;
  __syncthreads();
  for (int sz = 2; sz <= 256; sz <<= 1) {
    for (int str = sz >> 1; str > 0; str >>= 1) {
      int i = tid, j = i ^ str;
      if (j > i) {
        bool up = ((i & sz) == 0);
        float ki = key[i], kj = key[j];
        int vi = idx[i], vj = idx[j];
        bool gt = (ki > kj) || (ki == kj && vi > vj);
        if (gt == up) { key[i] = kj; key[j] = ki; idx[i] = vj; idx[j] = vi; }
      }
      __syncthreads();
    }
  }
  ORD[fb * 256 + tid] = idx[tid];
}

// ---------------------------------------------------------------------------
// Gather filt(scores/anchors/ctx_anchors): sorted ranks [192..255] then [0..63]
// ---------------------------------------------------------------------------
__global__ void gather_kernel(const float* __restrict__ SC, const int* __restrict__ ORD,
                              float* __restrict__ OUT) {
  const int fb = blockIdx.x, r = threadIdx.x;   // 12 blocks x 128 threads
  const int f = fb >> 2;
  const int K = 64;
  const int sr = (r < K) ? (192 + r) : (r - K);
  const int t = ORD[fb * 256 + sr];
  const int RFs[3] = {6, 12, 24};
  const int s = RFs[f];
  float* so_ = OUT + ((size_t)fb * 128 + r) * CC;
  const float* sp = SC + ((size_t)fb * TT + t) * CC;
  for (int c = 0; c < CC; ++c) so_[c] = sp[c];
  float* an = OUT + 30720 + ((size_t)fb * 128 + r) * 2;
  an[0] = (float)(t - (s / 2 - 1));
  an[1] = (float)(t + s / 2);
  float* can = OUT + 33792 + ((size_t)fb * 128 + r) * 2;
  can[0] = (float)(t - (s - 1));
  can[1] = (float)(t + s);
}

// ---------------------------------------------------------------------------
extern "C" void kernel_launch(void* const* d_in, const int* in_sizes, int n_in,
                              void* d_out, int out_size, void* d_ws, size_t ws_size,
                              hipStream_t stream) {
  (void)in_sizes; (void)n_in; (void)out_size; (void)ws_size;
  const float* x = (const float*)d_in[0];
  float* out = (float*)d_out;
  char* ws = (char*)d_ws;

  size_t off = 0;
  __bf16* w1b = (__bf16*)(ws + off); off += (size_t)3 * DD * DD * 2;        // 25 MB
  __bf16* w2b = (__bf16*)(ws + off); off += (size_t)3 * DD * DD * 2;        // 25 MB
  __bf16* hb  = (__bf16*)(ws + off); off += (size_t)4 * PT * DD * 2;        // 5.2 MB
  __bf16* y1b = (__bf16*)(ws + off); off += (size_t)4 * PT * DD * 2;        // 5.2 MB
  __bf16* lgb = (__bf16*)(ws + off); off += (size_t)4 * PT * DD * 2;        // 5.2 MB
  __bf16* cwb = (__bf16*)(ws + off); off += (size_t)32 * 48 * DD * 2;       // 6.3 MB
  float*  scb = (float*)(ws + off);  off += (size_t)3 * 4 * TT * CC * 4;
  int*    ordb = (int*)(ws + off);   off += (size_t)12 * 256 * 4;

  // zero hb/y1b/lgb (pads) and cwb (tail rows) once per launch; interiors are
  // fully rewritten each scale, pads never written -> deterministic
  {
    int nz = 3 * 4 * PT * DD + 32 * 48 * DD;   // contiguous region hb..cwb
    zero_bf16_kernel<<<(nz + 255) / 256, 256, 0, stream>>>(hb, nz);
  }

  const int RFs[3] = {6, 12, 24};

  for (int f = 0; f < 3; ++f) {
    const float* w1 = (const float*)d_in[1 + 6 * f + 0];
    const float* b1 = (const float*)d_in[1 + 6 * f + 1];
    const float* w2 = (const float*)d_in[1 + 6 * f + 2];
    const float* b2 = (const float*)d_in[1 + 6 * f + 3];
    const float* cw = (const float*)d_in[1 + 6 * f + 4];
    const float* cb = (const float*)d_in[1 + 6 * f + 5];
    const int s = RFs[f];
    const int pad = s / 6;
    const int d1 = 2 * pad + 1;        // conv1 dilation
    const int d2 = (s / 6) * 4 + 1;    // conv2 dilation

    convw_kernel<<<DD * DD / 256, 256, 0, stream>>>(w1, w1b);
    convw_kernel<<<DD * DD / 256, 256, 0, stream>>>(w2, w2b);
    const int ncw = CC * 2 * s * DD;
    convcw_kernel<<<(ncw + 255) / 256, 256, 0, stream>>>(cw, cwb, ncw);
    maxpool_kernel<<<4 * TT * DD / 256, 256, 0, stream>>>(x, hb, pad);

    dim3 gg(DD / 128, TT / 64, 4);
    conv_gemm_kernel<<<gg, 256, 0, stream>>>(w1b, hb, b1, d1, y1b, (float*)nullptr);
    float* lgf = out + 36864 + (size_t)f * 4 * TT * DD;
    conv_gemm_kernel<<<gg, 256, 0, stream>>>(w2b, y1b, b2, d2, lgb, lgf);

    float* scf = scb + (size_t)f * 4 * TT * CC;
    dim3 sg(TT / 128, 4);
    score_gemm_kernel<<<sg, 256, 0, stream>>>(lgb, cwb, cb, 2 * s, s - 1, scf);
  }

  sort_kernel<<<12, 256, 0, stream>>>(scb, ordb);
  gather_kernel<<<12, 128, 0, stream>>>(scb, ordb, out);
}
